// FeatureGrouper_46755013984988
// MI455X (gfx1250) — matrix operand order repaired and checked by reading, not compiled, not run on hardware
//
#include <hip/hip_runtime.h>
#include <hip/hip_bf16.h>

// Problem constants (fixed by setup_inputs)
#define B_SZ    8
#define N1_PTS  1024
#define N2_PTS  8192
#define C_DIM   256
#define KNN     50
#define ROWS_TOTAL (B_SZ * N1_PTS * KNN)   // 409600 rows, divisible by 32
#define BN_EPS  1e-5f

typedef __bf16 bf16_t;
typedef bf16_t v16bf __attribute__((ext_vector_type(16)));
typedef float  v8f   __attribute__((ext_vector_type(8)));
typedef int    v4i   __attribute__((ext_vector_type(4)));

// Async global->LDS path (CDNA5 GLOBAL_LOAD_ASYNC_TO_LDS_B128, ASYNCcnt):
// guarded so the file compiles identically on toolchains without the builtins.
#if defined(__has_builtin)
#if __has_builtin(__builtin_amdgcn_global_load_async_to_lds_b128) && \
    __has_builtin(__builtin_amdgcn_s_wait_asynccnt)
#define USE_ASYNC_LDS 1
#endif
#endif
#ifndef USE_ASYNC_LDS
#define USE_ASYNC_LDS 0
#endif

#if USE_ASYNC_LDS
typedef __attribute__((address_space(1))) v4i gv4i_t;   // global (AS1)
typedef __attribute__((address_space(3))) v4i lv4i_t;   // LDS (AS3)
#endif

__device__ __forceinline__ bf16_t f2bf(float x) {
  // round-to-nearest-even f32 -> bf16 via bit manipulation
  unsigned u = __float_as_uint(x);
  unsigned r = (u + 0x7FFFu + ((u >> 16) & 1u)) >> 16;
  unsigned short h = (unsigned short)r;
  return __builtin_bit_cast(bf16_t, h);
}

// ---------------------------------------------------------------------------
// Kernel 0: zero the stats scratch (sum/sumsq for both BN layers)
// ---------------------------------------------------------------------------
__global__ void zero_stats_kernel(float* __restrict__ p) {
  p[blockIdx.x * 256 + threadIdx.x] = 0.0f;
}

// ---------------------------------------------------------------------------
// Kernel 1: KNN. One block serves 8 queries of one batch.
// Whole batch xyz (8192x3 f32 = 96KB, interleaved) cached in dynamic LDS --
// CDNA5 WGP has 320KB LDS. The fill is a pure contiguous copy, so it uses
// GLOBAL_LOAD_ASYNC_TO_LDS_B128 (ASYNCcnt) when available. Then 50 rounds of
// argmin over a 64-bit (dist_bits << 32 | index) key: distances >= 0 so the
// f32 bit pattern orders as uint, and low index bits reproduce jax top_k's
// first-occurrence tie-breaking.
// ---------------------------------------------------------------------------
#define KNN_QPB 8
__global__ __launch_bounds__(256)
void knn_kernel(const float* __restrict__ xyz,
                const float* __restrict__ p1,
                int* __restrict__ idx_out) {
  extern __shared__ char smem_raw[];
  float* spts = (float*)smem_raw;                                  // [8192*3]
  unsigned char* flags = (unsigned char*)(spts + N2_PTS * 3);      // [8192]
  unsigned long long* red = (unsigned long long*)(flags + N2_PTS); // [256]

  const int tid = threadIdx.x;
  const int blocksPerBatch = N1_PTS / KNN_QPB;           // 128
  const int b = blockIdx.x / blocksPerBatch;
  const int qbase = (blockIdx.x % blocksPerBatch) * KNN_QPB;

  const float* gbase = xyz + (size_t)b * N2_PTS * 3;     // 96KB, 16B aligned

#if USE_ASYNC_LDS
  {
    gv4i_t* g = (gv4i_t*)gbase;
    lv4i_t* l = (lv4i_t*)spts;
    // 8192*3*4B / 16B = 6144 B128 transfers
    for (int j = tid; j < (N2_PTS * 3) / 4; j += 256) {
      __builtin_amdgcn_global_load_async_to_lds_b128(g + j, l + j, 0, 0);
    }
    __builtin_amdgcn_s_wait_asynccnt(0);
  }
#else
  {
    const float4* g = (const float4*)gbase;
    float4* l = (float4*)spts;
    for (int j = tid; j < (N2_PTS * 3) / 4; j += 256) l[j] = g[j];
  }
#endif
  __syncthreads();

  for (int q = 0; q < KNN_QPB; ++q) {
    const int n1 = qbase + q;
    const float* qp = p1 + ((size_t)b * N1_PTS + n1) * 3;
    const float qx = qp[0], qy = qp[1], qz = qp[2];

    for (int i = tid; i < N2_PTS; i += 256) flags[i] = 0;
    __syncthreads();

    for (int sel = 0; sel < KNN; ++sel) {
      unsigned long long best = ~0ull;
      for (int i = tid; i < N2_PTS; i += 256) {
        if (flags[i]) continue;
        float dx = qx - spts[3 * i + 0];
        float dy = qy - spts[3 * i + 1];
        float dz = qz - spts[3 * i + 2];
        float d2 = dx * dx + dy * dy + dz * dz;
        unsigned long long key =
            ((unsigned long long)__float_as_uint(d2) << 32) | (unsigned)i;
        best = (key < best) ? key : best;
      }
      red[tid] = best;
      __syncthreads();
      #pragma unroll
      for (int s = 128; s > 0; s >>= 1) {
        if (tid < s) {
          unsigned long long o = red[tid + s];
          if (o < red[tid]) red[tid] = o;
        }
        __syncthreads();
      }
      const int win = (int)(red[0] & 0xFFFFFFFFull);
      if (tid == 0) {
        idx_out[((size_t)b * N1_PTS + n1) * KNN + sel] = win;
        flags[win] = 1;
      }
      __syncthreads();
    }
  }
}

// ---------------------------------------------------------------------------
// Kernel 2/4: fused (gather | BN1+ReLU) + GEMM + bias + BN-stats.
// Block = 256 threads (8 wave32), computes a 32(row) x 64(channel) tile.
// Waves arranged 2x4; each wave does one 16x16 WMMA tile, K=256 in 8 steps
// of v_wmma_f32_16x16x32_bf16 (f32 accumulate).
// MODE 0: A rows gathered from feat via KNN idx (feat=67MB -> L2 hits).
// MODE 1: A rows = h1 with BN1 scale/shift + ReLU fused into the tile load.
// ---------------------------------------------------------------------------
template <int MODE>
__global__ __launch_bounds__(256)
void gemm_kernel(const float* __restrict__ Asrc,   // feat (MODE0) or h1 (MODE1)
                 const int*   __restrict__ idx,    // MODE0 only
                 const float* __restrict__ W,      // [256][256] row-major (o,c)
                 const float* __restrict__ bias,   // [256]
                 const float* __restrict__ bn_s,   // MODE1 only
                 const float* __restrict__ bn_t,   // MODE1 only
                 float* __restrict__ out,          // [ROWS][256]
                 float* __restrict__ gsum,         // [256]
                 float* __restrict__ gsq) {        // [256]
  __shared__ bf16_t As[32][C_DIM + 8];
  __shared__ bf16_t Bs[64][C_DIM + 8];
  __shared__ float  s_sum[64];
  __shared__ float  s_sq[64];

  const int tid = threadIdx.x;
  const int r0 = blockIdx.x * 32;
  const int c0 = blockIdx.y * 64;

  if (tid < 64) { s_sum[tid] = 0.0f; s_sq[tid] = 0.0f; }

  // ---- A tile: 32 rows x 256 (8 threads/row, 32 floats each, float4 loads)
  {
    const int row = tid >> 3;
    const int seg = tid & 7;
    const int r = r0 + row;
    const float* src;
    if (MODE == 0) {
      const int b = r / (N1_PTS * KNN);
      const int s = idx[r];
      src = Asrc + ((size_t)b * N2_PTS + s) * C_DIM;
    } else {
      src = Asrc + (size_t)r * C_DIM;
    }
    const float4* src4 = (const float4*)src + seg * 8;
    #pragma unroll
    for (int j = 0; j < 8; ++j) {
      float4 v = src4[j];
      const int c = seg * 32 + j * 4;
      if (MODE == 1) {
        v.x = fmaxf(0.0f, v.x * bn_s[c + 0] + bn_t[c + 0]);
        v.y = fmaxf(0.0f, v.y * bn_s[c + 1] + bn_t[c + 1]);
        v.z = fmaxf(0.0f, v.z * bn_s[c + 2] + bn_t[c + 2]);
        v.w = fmaxf(0.0f, v.w * bn_s[c + 3] + bn_t[c + 3]);
      }
      As[row][c + 0] = f2bf(v.x);
      As[row][c + 1] = f2bf(v.y);
      As[row][c + 2] = f2bf(v.z);
      As[row][c + 3] = f2bf(v.w);
    }
  }

  // ---- B tile: W rows [c0, c0+64), all K=256 (4 threads/row, 64 floats each)
  {
    const int n = tid >> 2;
    const int seg = tid & 3;
    const float4* src4 = (const float4*)(W + (size_t)(c0 + n) * C_DIM) + seg * 16;
    #pragma unroll
    for (int j = 0; j < 16; ++j) {
      float4 v = src4[j];
      const int k = seg * 64 + j * 4;
      Bs[n][k + 0] = f2bf(v.x);
      Bs[n][k + 1] = f2bf(v.y);
      Bs[n][k + 2] = f2bf(v.z);
      Bs[n][k + 3] = f2bf(v.w);
    }
  }
  __syncthreads();

  // ---- WMMA main loop
  const int lane = tid & 31;
  const int wid  = tid >> 5;
  const int wr   = wid & 1;    // row sub-tile 0..1
  const int wc   = wid >> 1;   // col sub-tile 0..3
  const int m    = lane & 15;
  const int hi   = lane >> 4;  // lane half (0/1)
  const int arow = wr * 16 + m;
  const int ncol = wc * 16 + m;

  v8f acc = {};
  #pragma unroll
  for (int kc = 0; kc < C_DIM / 32; ++kc) {
    const int k0 = kc * 32;
    v16bf a, bm;
    #pragma unroll
    for (int v = 0; v < 8; ++v) {
      // A 16x32 bf16 fragment layout (ISA 7.12.2)
      const int ka = k0 + ((v >= 4) ? 16 : 0) + hi * 8 + (v & 3) * 2;
      a[2 * v]     = As[arow][ka];
      a[2 * v + 1] = As[arow][ka + 1];
      // B 32x16 bf16 fragment layout
      const int kb = k0 + ((v >= 4) ? 16 : 0) + hi * 8 + (v & 3) * 2;
      bm[2 * v]     = Bs[ncol][kb];
      bm[2 * v + 1] = Bs[ncol][kb + 1];
    }
    acc = __builtin_amdgcn_wmma_f32_16x16x32_bf16(
        false, a, false, bm, (short)0, acc, false, false);
  }

  // ---- epilogue: +bias, store, per-channel sum / sumsq reduction
  const int n = c0 + ncol;
  const float bv = bias[n];
  float sl = 0.0f, ql = 0.0f;
  #pragma unroll
  for (int v = 0; v < 8; ++v) {
    const int row = r0 + wr * 16 + v + hi * 8;   // D layout: M = v + 8*hi
    const float val = acc[v] + bv;
    out[(size_t)row * C_DIM + n] = val;
    sl += val;
    ql += val * val;
  }
  // combine the two lane halves (same channel, different rows)
  sl += __shfl_xor(sl, 16);
  ql += __shfl_xor(ql, 16);
  if (lane < 16) {
    atomicAdd(&s_sum[wc * 16 + lane], sl);
    atomicAdd(&s_sq [wc * 16 + lane], ql);
  }
  __syncthreads();
  if (tid < 64) {
    atomicAdd(&gsum[c0 + tid], s_sum[tid]);
    atomicAdd(&gsq [c0 + tid], s_sq [tid]);
  }
}

// ---------------------------------------------------------------------------
// Kernel 3/5: fold BN stats into per-channel scale/shift: y = s*x + t
// ---------------------------------------------------------------------------
__global__ void finalize_stats_kernel(const float* __restrict__ sum,
                                      const float* __restrict__ sq,
                                      const float* __restrict__ gamma,
                                      const float* __restrict__ beta,
                                      float* __restrict__ s,
                                      float* __restrict__ t) {
  const int c = threadIdx.x;
  const float inv_n = 1.0f / (float)ROWS_TOTAL;
  const float m  = sum[c] * inv_n;
  const float vr = sq[c] * inv_n - m * m;
  const float r  = rsqrtf(vr + BN_EPS);
  const float sc = gamma[c] * r;
  s[c] = sc;
  t[c] = beta[c] - m * sc;
}

// ---------------------------------------------------------------------------
// Kernel 6: apply BN2 in place on the output
// ---------------------------------------------------------------------------
__global__ __launch_bounds__(256)
void bn_apply_kernel(float* __restrict__ out,
                     const float* __restrict__ s,
                     const float* __restrict__ t) {
  const size_t total = (size_t)ROWS_TOTAL * C_DIM;
  for (size_t i = (size_t)blockIdx.x * blockDim.x + threadIdx.x; i < total;
       i += (size_t)gridDim.x * blockDim.x) {
    const int c = (int)(i & (C_DIM - 1));
    out[i] = s[c] * out[i] + t[c];
  }
}

// ---------------------------------------------------------------------------
extern "C" void kernel_launch(void* const* d_in, const int* in_sizes, int n_in,
                              void* d_out, int out_size, void* d_ws, size_t ws_size,
                              hipStream_t stream) {
  (void)in_sizes; (void)n_in; (void)out_size; (void)ws_size;
  const float* xyz  = (const float*)d_in[0];
  const float* p1   = (const float*)d_in[1];
  const float* feat = (const float*)d_in[2];
  const float* W1   = (const float*)d_in[3];
  const float* b1   = (const float*)d_in[4];
  const float* g1   = (const float*)d_in[5];
  const float* be1  = (const float*)d_in[6];
  const float* W2   = (const float*)d_in[7];
  const float* b2   = (const float*)d_in[8];
  const float* g2   = (const float*)d_in[9];
  const float* be2  = (const float*)d_in[10];
  float* out = (float*)d_out;

  // workspace layout: [8x256 stats][409600 idx][409600x256 h1]
  float* wsf  = (float*)d_ws;
  float* sum1 = wsf + 0 * 256;
  float* sq1  = wsf + 1 * 256;
  float* s1   = wsf + 2 * 256;
  float* t1   = wsf + 3 * 256;
  float* sum2 = wsf + 4 * 256;
  float* sq2  = wsf + 5 * 256;
  float* s2   = wsf + 6 * 256;
  float* t2   = wsf + 7 * 256;
  int*   idx  = (int*)(wsf + 2048);
  float* h1   = wsf + 2048 + ROWS_TOTAL;   // 16B-aligned offset

  zero_stats_kernel<<<8, 256, 0, stream>>>(wsf);

  const size_t knn_lds = (size_t)N2_PTS * 3 * sizeof(float)   // xyz cache
                       + (size_t)N2_PTS                       // flags
                       + 256 * sizeof(unsigned long long);    // reduction
  knn_kernel<<<B_SZ * (N1_PTS / KNN_QPB), 256, knn_lds, stream>>>(xyz, p1, idx);

  dim3 ggrid(ROWS_TOTAL / 32, C_DIM / 64);
  gemm_kernel<0><<<ggrid, 256, 0, stream>>>(feat, idx, W1, b1, nullptr, nullptr,
                                            h1, sum1, sq1);
  finalize_stats_kernel<<<1, 256, 0, stream>>>(sum1, sq1, g1, be1, s1, t1);
  gemm_kernel<1><<<ggrid, 256, 0, stream>>>(h1, nullptr, W2, b2, s1, t1,
                                            out, sum2, sq2);
  finalize_stats_kernel<<<1, 256, 0, stream>>>(sum2, sq2, g2, be2, s2, t2);
  bn_apply_kernel<<<4096, 256, 0, stream>>>(out, s2, t2);
}
